// CoreGNN_6966436954273
// MI455X (gfx1250) — compile-verified, matrix operand order checked by reading, not run on hardware
//
#include <hip/hip_runtime.h>
#include <math.h>

// ---------------------------------------------------------------------------
// Problem constants (from reference setup_inputs)
// ---------------------------------------------------------------------------
#define NN   20000      // nodes
#define EE   160000     // edges
#define GG   400        // graphs
#define DM   128        // input / hidden dim (D == HD == 128)
#define FF   512        // H * HD
#define HH   4          // heads
#define HDI  128        // per-head dim
#define EDI  16         // edge feature dim

typedef __attribute__((ext_vector_type(16))) __bf16        v16bf;
typedef __attribute__((ext_vector_type(8)))  float         v8f;
typedef __attribute__((ext_vector_type(8)))  unsigned short u16x8;

union BF16x16 { v16bf v; u16x8 h[2]; unsigned short u[16]; };

// WMMA fragment element -> K offset within a 32-wide k-step
__device__ __host__ __forceinline__ int frag_k(int lane, int e) {
  return ((lane >> 4) * 8) + e + ((e >= 8) ? 8 : 0);
}

// ---------------------------------------------------------------------------
// Pack A[M,K] (f32, row-major) -> Apack (bf16, WMMA A-fragment order)
// flat = ((tm*KS + ks)*32 + lane)*16 + e
// ---------------------------------------------------------------------------
__global__ void k_pack_A(const float* __restrict__ A, unsigned short* __restrict__ Ap,
                         int M, int K) {
  int t = blockIdx.x * blockDim.x + threadIdx.x;
  if (t >= M * K) return;
  const int KS = K >> 5;
  int e = t & 15, w = t >> 4;
  int lane = w & 31; w >>= 5;
  int ks = w % KS, tm = w / KS;
  int m  = tm * 16 + (lane & 15);
  int kk = ks * 32 + frag_k(lane, e);
  Ap[t] = (unsigned short)(__float_as_uint(A[(size_t)m * K + kk]) >> 16);
}

// Pack B[K,Ncols] (f32, row-major) -> Bpack (bf16, WMMA B-fragment order)
// flat = ((tn*KS + ks)*32 + lane)*16 + e
__global__ void k_pack_B(const float* __restrict__ B, unsigned short* __restrict__ Bp,
                         int K, int Ncols) {
  int t = blockIdx.x * blockDim.x + threadIdx.x;
  if (t >= K * Ncols) return;
  const int KS = K >> 5;
  int e = t & 15, w = t >> 4;
  int lane = w & 31; w >>= 5;
  int ks = w % KS, tn = w / KS;
  int n  = tn * 16 + (lane & 15);
  int kk = ks * 32 + frag_k(lane, e);
  Bp[t] = (unsigned short)(__float_as_uint(B[(size_t)kk * Ncols + n]) >> 16);
}

// ---------------------------------------------------------------------------
// WMMA bf16 GEMM on pre-packed fragments:
//   C[M,Ncols] = A @ B (+bias, optional relu)
// One wave -> one 16x16 tile; per k-step: 2+2 global_load_b128 + 1 v_wmma.
// ---------------------------------------------------------------------------
__global__ void wmma_gemm_pk(const unsigned short* __restrict__ Ap,
                             const unsigned short* __restrict__ Bp,
                             float* __restrict__ C,
                             int M, int K, int Ncols,
                             const float* __restrict__ bias, int do_relu) {
  const int gtid = blockIdx.x * blockDim.x + threadIdx.x;
  const int wave = gtid >> 5;
  const int lane = threadIdx.x & 31;
  const int ntn  = Ncols >> 4;
  const int tm   = wave / ntn;          // wave-uniform
  const int tn   = wave % ntn;
  if (tm * 16 >= M) return;             // uniform per wave -> EXEC stays all-1
  const int KS = K >> 5;

  const u16x8* ap = (const u16x8*)(Ap + (((size_t)tm * KS) * 32 + lane) * 16);
  const u16x8* bp = (const u16x8*)(Bp + (((size_t)tn * KS) * 32 + lane) * 16);

  v8f acc = {};
  for (int ks = 0; ks < KS; ++ks) {
    BF16x16 a, b;
    a.h[0] = ap[0]; a.h[1] = ap[1];     // 2x global_load_b128, contiguous
    b.h[0] = bp[0]; b.h[1] = bp[1];
    acc = __builtin_amdgcn_wmma_f32_16x16x32_bf16(
        false, a.v, false, b.v, (short)0, acc, false, false);
    ap += 64;                           // 32 lanes * 16 bf16 / 8 = 64 u16x8
    bp += 64;
  }

  const int ncol  = tn * 16 + (lane & 15);
  const int mbase = tm * 16 + ((lane >> 4) * 8);
  const float bv  = bias ? bias[ncol] : 0.f;
#pragma unroll
  for (int r = 0; r < 8; ++r) {
    float v = acc[r] + bv;
    if (do_relu) v = fmaxf(v, 0.f);
    C[(size_t)(mbase + r) * Ncols + ncol] = v;
  }
}

// ---------------------------------------------------------------------------
// Small helper kernels
// ---------------------------------------------------------------------------
__global__ void k_fill(float* p, float v, int n) {
  int i = blockIdx.x * blockDim.x + threadIdx.x;
  if (i < n) p[i] = v;
}

__global__ void k_count(const int* __restrict__ batch, float* cnt, int n) {
  int i = blockIdx.x * blockDim.x + threadIdx.x;
  if (i < n) atomicAdd(&cnt[batch[i]], 1.f);
}

// fold We ([16,512]) with a_e ([4,128]) -> w_comb[4,16]
__global__ void k_fold_we(const float* __restrict__ We,
                          const float* __restrict__ ae,
                          float* __restrict__ wc) {
  int t = threadIdx.x;                  // 0..63
  int h = t >> 4, d = t & 15;
  float s = 0.f;
  for (int c = 0; c < HDI; ++c) s += We[d * FF + h * HDI + c] * ae[h * HDI + c];
  wc[h * 16 + d] = s;
}

// per-(node,head) attention scores: one wave per (n,h)
__global__ void k_node_scores(const float* __restrict__ hf,
                              const float* __restrict__ asrc,
                              const float* __restrict__ adst,
                              float* __restrict__ ssrc,
                              float* __restrict__ sdst, int n) {
  int gtid = blockIdx.x * blockDim.x + threadIdx.x;
  int wave = gtid >> 5, lane = gtid & 31;
  int node = wave >> 2, h = wave & 3;
  if (node >= n) return;
  const float* hp = hf + (size_t)node * FF + h * HDI;
  float ss = 0.f, sd = 0.f;
  for (int c = lane; c < HDI; c += 32) {
    float v = hp[c];
    ss += v * asrc[h * HDI + c];
    sd += v * adst[h * HDI + c];
  }
#pragma unroll
  for (int off = 16; off; off >>= 1) {
    ss += __shfl_down(ss, off);
    sd += __shfl_down(sd, off);
  }
  if (lane == 0) { ssrc[node * HH + h] = ss; sdst[node * HH + h] = sd; }
}

__device__ __forceinline__ void atomicMaxF(float* addr, float val) {
  if (val >= 0.f) atomicMax((int*)addr, __float_as_int(val));
  else            atomicMin((unsigned int*)addr, __float_as_uint(val));
}

__global__ void k_edge_logits(const float* __restrict__ ea,
                              const int* __restrict__ ei,
                              const float* __restrict__ wc,
                              const float* __restrict__ ssrc,
                              const float* __restrict__ sdst,
                              float* __restrict__ al,
                              float* __restrict__ mx, int e_cnt) {
  int e = blockIdx.x * blockDim.x + threadIdx.x;
  if (e >= e_cnt) return;
  int s = ei[e], d = ei[e_cnt + e];
  float ef[EDI];
#pragma unroll
  for (int j = 0; j < EDI; ++j) ef[j] = ea[(size_t)e * EDI + j];
#pragma unroll
  for (int h = 0; h < HH; ++h) {
    float dot = 0.f;
#pragma unroll
    for (int j = 0; j < EDI; ++j) dot += ef[j] * wc[h * 16 + j];
    float a = ssrc[s * HH + h] + sdst[d * HH + h] + dot;
    a = a > 0.f ? a : 0.2f * a;         // leaky_relu(0.2)
    al[(size_t)e * HH + h] = a;
    atomicMaxF(&mx[d * HH + h], a);
  }
}

__global__ void k_mfix(float* m, int n) {
  int i = blockIdx.x * blockDim.x + threadIdx.x;
  if (i < n) { float v = m[i]; if (!(v > -1e38f && v < 1e38f)) m[i] = 0.f; }
}

__global__ void k_softmax_p(const int* __restrict__ ei,
                            float* __restrict__ al,
                            const float* __restrict__ mx,
                            float* __restrict__ den, int e_cnt) {
  int e = blockIdx.x * blockDim.x + threadIdx.x;
  if (e >= e_cnt) return;
  int d = ei[e_cnt + e];
#pragma unroll
  for (int h = 0; h < HH; ++h) {
    float p = __expf(al[(size_t)e * HH + h] - mx[d * HH + h]);
    al[(size_t)e * HH + h] = p;
    atomicAdd(&den[d * HH + h], p);
  }
}

// message aggregation: out[dst, :512] += h[src, :512] * alpha[e, head]
__global__ void k_aggregate(const int* __restrict__ ei,
                            const float* __restrict__ hf,
                            const float* __restrict__ p,
                            const float* __restrict__ den,
                            float* __restrict__ outa, int e_cnt) {
  int gid = blockIdx.x * blockDim.x + threadIdx.x;
  if (gid >= e_cnt * FF) return;
  int e = gid >> 9, idx = gid & (FF - 1), h = idx >> 7;
  int s = ei[e], d = ei[e_cnt + e];
  float alpha = p[(size_t)e * HH + h] / (den[d * HH + h] + 1e-16f);
  atomicAdd(&outa[(size_t)d * FF + idx], hf[(size_t)s * FF + idx] * alpha);
}

__global__ void k_headmean(const float* __restrict__ agg,
                           const float* __restrict__ b,
                           float* __restrict__ hb, int n) {
  int i = blockIdx.x * blockDim.x + threadIdx.x;
  if (i >= n) return;
  int node = i >> 7, c = i & 127;
  const float* a = agg + (size_t)node * FF;
  hb[i] = 0.25f * (a[c] + a[HDI + c] + a[2 * HDI + c] + a[3 * HDI + c]) + b[c];
}

// ------------------------- GraphNorm ---------------------------------------
__global__ void k_seg_accum(const float* __restrict__ h,
                            const int* __restrict__ batch,
                            float* __restrict__ gsum, int n) {
  int i = blockIdx.x * blockDim.x + threadIdx.x;
  if (i >= n) return;
  int node = i >> 7, c = i & 127;
  atomicAdd(&gsum[batch[node] * DM + c], h[i]);
}

__global__ void k_seg_div(float* gsum, const float* __restrict__ cnt, int n) {
  int i = blockIdx.x * blockDim.x + threadIdx.x;
  if (i < n) gsum[i] /= fmaxf(cnt[i >> 7], 1.f);
}

__global__ void k_gn_var(const float* __restrict__ h,
                         const int* __restrict__ batch,
                         const float* __restrict__ mean,
                         const float* __restrict__ ms,
                         float* __restrict__ gvar, int n) {
  int i = blockIdx.x * blockDim.x + threadIdx.x;
  if (i >= n) return;
  int node = i >> 7, c = i & 127, g = batch[node];
  float xc = h[i] - ms[c] * mean[g * DM + c];
  atomicAdd(&gvar[g * DM + c], xc * xc);
}

__global__ void k_gn_final(float* __restrict__ h,
                           const int* __restrict__ batch,
                           const float* __restrict__ mean,
                           const float* __restrict__ gvar,
                           const float* __restrict__ cnt,
                           const float* __restrict__ ms,
                           const float* __restrict__ w,
                           const float* __restrict__ b, int n) {
  int i = blockIdx.x * blockDim.x + threadIdx.x;
  if (i >= n) return;
  int node = i >> 7, c = i & 127, g = batch[node];
  float xc  = h[i] - ms[c] * mean[g * DM + c];
  float var = gvar[g * DM + c] / fmaxf(cnt[g], 1.f);
  float y   = w[c] * xc * rsqrtf(var + 1e-5f) + b[c];
  h[i] = fmaxf(y, 0.f);                 // fused trailing ReLU of the layer
}

// readout layer 2: out[g] = r1[g,:64] @ W2 + b2
__global__ void k_ro2(const float* __restrict__ r1,
                      const float* __restrict__ W2,
                      const float* __restrict__ b2,
                      float* __restrict__ out, int g_cnt, int k) {
  int g = blockIdx.x * blockDim.x + threadIdx.x;
  if (g >= g_cnt) return;
  float s = 0.f;
  for (int j = 0; j < k; ++j) s += r1[g * k + j] * W2[j];
  out[g] = s + b2[0];
}

// ---------------------------------------------------------------------------
// Host side
// ---------------------------------------------------------------------------
static inline int cdiv(int a, int b) { return (a + b - 1) / b; }

extern "C" void kernel_launch(void* const* d_in, const int* in_sizes, int n_in,
                              void* d_out, int out_size, void* d_ws, size_t ws_size,
                              hipStream_t stream) {
  (void)in_sizes; (void)n_in; (void)out_size; (void)ws_size;

  const float* x        = (const float*)d_in[0];
  const float* eattr    = (const float*)d_in[1];
  const float* ceattr   = (const float*)d_in[2];
  const float* enc_W    = (const float*)d_in[3];
  const float* enc_as   = (const float*)d_in[4];
  const float* enc_ad   = (const float*)d_in[5];
  const float* enc_We   = (const float*)d_in[6];
  const float* enc_ae   = (const float*)d_in[7];
  const float* enc_b    = (const float*)d_in[8];
  const float* enc_nw   = (const float*)d_in[9];
  const float* enc_nb   = (const float*)d_in[10];
  const float* enc_nms  = (const float*)d_in[11];
  const float* cr_W     = (const float*)d_in[12];
  const float* cr_as    = (const float*)d_in[13];
  const float* cr_ad    = (const float*)d_in[14];
  const float* cr_We    = (const float*)d_in[15];
  const float* cr_ae    = (const float*)d_in[16];
  const float* cr_b     = (const float*)d_in[17];
  const float* cr_nw    = (const float*)d_in[18];
  const float* cr_nb    = (const float*)d_in[19];
  const float* cr_nms   = (const float*)d_in[20];
  const float* ro_W1    = (const float*)d_in[21];
  const float* ro_b1    = (const float*)d_in[22];
  const float* ro_W2    = (const float*)d_in[23];
  const float* ro_b2    = (const float*)d_in[24];
  const int*   eidx     = (const int*)d_in[25];
  const int*   ceidx    = (const int*)d_in[26];
  const int*   batch    = (const int*)d_in[27];
  float*       out      = (float*)d_out;

  // ---- workspace carve-up ----
  float* ws = (float*)d_ws;
  size_t o = 0;
  float* h_buf   = ws + o; o += (size_t)NN * DM;   // [N,128] activations
  float* h_feat  = ws + o; o += (size_t)NN * FF;   // [N,512] h = X@W
  float* out_agg = ws + o; o += (size_t)NN * FF;   // [N,512] aggregated messages
  float* s_src   = ws + o; o += (size_t)NN * HH;
  float* s_dst   = ws + o; o += (size_t)NN * HH;
  float* al      = ws + o; o += (size_t)EE * HH;   // logits -> reused as p
  float* mx      = ws + o; o += (size_t)NN * HH;
  float* den     = ws + o; o += (size_t)NN * HH;
  float* w_comb  = ws + o; o += 64;
  float* gsum    = ws + o; o += (size_t)GG * DM;
  float* gvar    = ws + o; o += (size_t)GG * DM;
  float* cnt     = ws + o; o += GG;
  float* pooled  = ws + o; o += (size_t)GG * DM;
  float* r1      = ws + o; o += (size_t)GG * 64;
  // packed bf16 fragment buffers (ushort), carved after the float region
  unsigned short* Apack = (unsigned short*)(ws + o);          // up to N*128 bf16
  unsigned short* Bpack = Apack + (size_t)NN * DM;            // up to 128*512 bf16

  const int B = 256;
  const float NEG_INF = -__builtin_inff();

  // per-graph node counts (batch is constant across layers)
  k_fill<<<cdiv(GG, B), B, 0, stream>>>(cnt, 0.f, GG);
  k_count<<<cdiv(NN, B), B, 0, stream>>>(batch, cnt, NN);

  const float* hin = x;
  for (int L = 0; L < 6; ++L) {
    const int  i   = L % 3;
    const bool enc = L < 3;
    const float* W   = (enc ? enc_W  : cr_W)  + (size_t)i * DM * FF;
    const float* asr = (enc ? enc_as : cr_as) + (size_t)i * HH * HDI;
    const float* adt = (enc ? enc_ad : cr_ad) + (size_t)i * HH * HDI;
    const float* Wei = (enc ? enc_We : cr_We) + (size_t)i * EDI * FF;
    const float* aei = (enc ? enc_ae : cr_ae) + (size_t)i * HH * HDI;
    const float* bi  = (enc ? enc_b  : cr_b)  + (size_t)i * HDI;
    const float* nwi = (enc ? enc_nw : cr_nw) + (size_t)i * HDI;
    const float* nbi = (enc ? enc_nb : cr_nb) + (size_t)i * HDI;
    const float* nms = (enc ? enc_nms: cr_nms)+ (size_t)i * HDI;
    const int*   ei  = enc ? eidx  : ceidx;
    const float* ea  = enc ? eattr : ceattr;

    // 1) pack operands to bf16 WMMA-fragment order, then GEMM via WMMA:
    //    h_feat = hin @ W   ([N,128]@[128,512])
    k_pack_A<<<cdiv(NN * DM, B), B, 0, stream>>>(hin, Apack, NN, DM);
    k_pack_B<<<cdiv(DM * FF, B), B, 0, stream>>>(W, Bpack, DM, FF);
    {
      int waves = (NN / 16) * (FF / 16);                 // 40000
      wmma_gemm_pk<<<cdiv(waves * 32, B), B, 0, stream>>>(
          Apack, Bpack, h_feat, NN, DM, FF, nullptr, 0);
    }

    // 2) fold edge-weight with a_e (kills the [E,512] intermediate)
    k_fold_we<<<1, 64, 0, stream>>>(Wei, aei, w_comb);

    // 3) per-node attention scores (wave per (n,h))
    k_node_scores<<<cdiv(NN * HH * 32, B), B, 0, stream>>>(
        h_feat, asr, adt, s_src, s_dst, NN);

    // 4) init reductions
    k_fill<<<cdiv(NN * HH, B), B, 0, stream>>>(mx, NEG_INF, NN * HH);
    k_fill<<<cdiv(NN * HH, B), B, 0, stream>>>(den, 0.f, NN * HH);
    k_fill<<<cdiv(NN * FF, B), B, 0, stream>>>(out_agg, 0.f, NN * FF);

    // 5) logits + leaky relu + segment max
    k_edge_logits<<<cdiv(EE, B), B, 0, stream>>>(
        ea, ei, w_comb, s_src, s_dst, al, mx, EE);
    k_mfix<<<cdiv(NN * HH, B), B, 0, stream>>>(mx, NN * HH);

    // 6) p = exp(al - m[dst]); den = segment_sum(p)
    k_softmax_p<<<cdiv(EE, B), B, 0, stream>>>(ei, al, mx, den, EE);

    // 7) message aggregation (the traffic-dominant scatter)
    k_aggregate<<<EE * (FF / B), B, 0, stream>>>(
        ei, h_feat, al, den, out_agg, EE);

    // 8) head mean + bias -> h_buf [N,128]
    k_headmean<<<cdiv(NN * DM, B), B, 0, stream>>>(out_agg, bi, h_buf, NN * DM);

    // 9) GraphNorm + ReLU (in place on h_buf)
    k_fill<<<cdiv(GG * DM, B), B, 0, stream>>>(gsum, 0.f, GG * DM);
    k_fill<<<cdiv(GG * DM, B), B, 0, stream>>>(gvar, 0.f, GG * DM);
    k_seg_accum<<<cdiv(NN * DM, B), B, 0, stream>>>(h_buf, batch, gsum, NN * DM);
    k_seg_div<<<cdiv(GG * DM, B), B, 0, stream>>>(gsum, cnt, GG * DM);
    k_gn_var<<<cdiv(NN * DM, B), B, 0, stream>>>(h_buf, batch, gsum, nms, gvar, NN * DM);
    k_gn_final<<<cdiv(NN * DM, B), B, 0, stream>>>(
        h_buf, batch, gsum, gvar, cnt, nms, nwi, nbi, NN * DM);

    hin = h_buf;
  }

  // ---- global mean pool ----
  k_fill<<<cdiv(GG * DM, B), B, 0, stream>>>(pooled, 0.f, GG * DM);
  k_seg_accum<<<cdiv(NN * DM, B), B, 0, stream>>>(h_buf, batch, pooled, NN * DM);
  k_seg_div<<<cdiv(GG * DM, B), B, 0, stream>>>(pooled, cnt, GG * DM);

  // ---- readout MLP (layer 1 via WMMA) ----
  k_pack_A<<<cdiv(GG * DM, B), B, 0, stream>>>(pooled, Apack, GG, DM);
  k_pack_B<<<cdiv(DM * 64, B), B, 0, stream>>>(ro_W1, Bpack, DM, 64);
  {
    int waves = (GG / 16) * (64 / 16);                    // 100 waves
    wmma_gemm_pk<<<cdiv(waves * 32, B), B, 0, stream>>>(
        Apack, Bpack, r1, GG, DM, 64, ro_b1, 1);
  }
  k_ro2<<<cdiv(GG, B), B, 0, stream>>>(r1, ro_W2, ro_b2, out, GG, 64);
}
